// SingleLayerRNN_11982958756767
// MI455X (gfx1250) — matrix operand order repaired and checked by reading, not compiled, not run on hardware
//
#include <hip/hip_runtime.h>
#include <hip/hip_bf16.h>
#include <stdint.h>

// ---------------------------------------------------------------------------
// CDNA5 WMMA bf16 path (probe-confirmed signature):
//   v8f = __builtin_amdgcn_wmma_f32_16x16x32_bf16(bool, v16bf, bool, v16bf,
//                                                 short, v8f, bool, bool)
// ---------------------------------------------------------------------------
typedef __attribute__((ext_vector_type(16))) __bf16 v16bf;
typedef __attribute__((ext_vector_type(8)))  float  v8f;

union ABPack {
    v16bf        v;
    uint4        q[2];
    unsigned int u32[8];
};
static_assert(sizeof(ABPack) == 32, "ABPack must be 32 bytes");

__device__ __forceinline__ unsigned short f2bf(float f) {
    // round-to-nearest-even fp32 -> bf16
    unsigned int u = __float_as_uint(f);
    u += 0x7FFFu + ((u >> 16) & 1u);
    return (unsigned short)(u >> 16);
}
__device__ __forceinline__ float bf2f(unsigned short u) {
    return __uint_as_float(((unsigned int)u) << 16);
}

#define TSEQ 512
#define BDIM 64
#define HDIM 1024
#define APAD 8
#define ASTRIDE (HDIM + APAD)   // 1032 bf16 elems; row = 2064 B (16B aligned)
#define MROWS 32                // A panel rows per block in phase 1
#define NBLK 256                // persistent-kernel grid (4 Mtiles x 64 Ntiles)

// ---------------------------------------------------------------------------
// fp32 -> bf16 weight conversion (2 elements / thread, packed dword stores)
// ---------------------------------------------------------------------------
__global__ void rnn_cvt_bf16_kernel(const float* __restrict__ src,
                                    unsigned short* __restrict__ dst, int n) {
    int i = (blockIdx.x * blockDim.x + threadIdx.x) * 2;
    if (i < n) {
        unsigned int lo = f2bf(src[i]);
        unsigned int hi = f2bf(src[i + 1]);
        *(unsigned int*)(dst + i) = lo | (hi << 16);
    }
}

__global__ void rnn_zero_kernel(unsigned int* __restrict__ p, int n) {
    int i = blockIdx.x * blockDim.x + threadIdx.x;
    if (i < n) p[i] = 0u;
}

// ---------------------------------------------------------------------------
// Phase 1: XW[32768,1024](bf16) = X[32768,1024](fp32) @ W_ih[1024,1024](bf16)
// (unchanged from previous round: LDS-staged A panel, 32x32 slab per wave,
//  each B-tile pair feeds two v_wmma; partial waits confirmed in disasm)
// ---------------------------------------------------------------------------
__global__ __launch_bounds__(256)
void rnn_xw_gemm_kernel(const float* __restrict__ X,
                        const unsigned short* __restrict__ Wih,
                        unsigned short* __restrict__ XW) {
    __shared__ unsigned short As[MROWS * ASTRIDE];   // 66,048 B

    const int tid    = threadIdx.x;
    const int lane   = tid & 31;
    const int wave   = tid >> 5;
    const int mpanel = blockIdx.x >> 2;              // 0..1023
    const int n0     = (blockIdx.x & 3) * 256 + wave * 32;

    {
        const float* src = X + (long)mpanel * MROWS * HDIM;
        unsigned int* dst = (unsigned int*)As;
        for (int q = tid; q < MROWS * HDIM / 4; q += 256) {
            int row = q >> 8;
            int c4  = q & 255;
            float4 f = *(const float4*)(src + row * HDIM + c4 * 4);
            unsigned int d0 = (unsigned)f2bf(f.x) | ((unsigned)f2bf(f.y) << 16);
            unsigned int d1 = (unsigned)f2bf(f.z) | ((unsigned)f2bf(f.w) << 16);
            uint2 d = {d0, d1};
            *(uint2*)(dst + row * (ASTRIDE / 2) + c4 * 2) = d;
        }
    }
    __syncthreads();

    const int half = lane >> 4;
    const int mrow = lane & 15;
    const unsigned short* a_base0 = As + mrow * ASTRIDE + half * 8;
    const unsigned short* a_base1 = a_base0 + 16 * ASTRIDE;
    const unsigned short* brow    = Wih + (long)(half * 8) * HDIM + n0 + mrow;

    v8f c[2][2] = {};

    for (int k0 = 0; k0 < HDIM; k0 += 32) {
        ABPack a0, a1;
        a0.q[0] = *(const uint4*)(a_base0 + k0);
        a0.q[1] = *(const uint4*)(a_base0 + k0 + 16);
        a1.q[0] = *(const uint4*)(a_base1 + k0);
        a1.q[1] = *(const uint4*)(a_base1 + k0 + 16);
        const unsigned short* bk = brow + (long)k0 * HDIM;
#pragma unroll
        for (int j = 0; j < 2; ++j) {
            ABPack b;
#pragma unroll
            for (int p = 0; p < 8; ++p)
                b.u32[p] = (unsigned)bk[(long)(2 * p + (p >= 4 ? 8 : 0)) * HDIM + j * 16] |
                           ((unsigned)bk[(long)(2 * p + 1 + (p >= 4 ? 8 : 0)) * HDIM + j * 16] << 16);
            c[0][j] = __builtin_amdgcn_wmma_f32_16x16x32_bf16(
                false, a0.v, false, b.v, (short)0, c[0][j], false, false);
            c[1][j] = __builtin_amdgcn_wmma_f32_16x16x32_bf16(
                false, a1.v, false, b.v, (short)0, c[1][j], false, false);
        }
    }

    const int ncol = lane & 15;
#pragma unroll
    for (int i = 0; i < 2; ++i) {
        const int mbase = mpanel * MROWS + i * 16 + half * 8;
#pragma unroll
        for (int j = 0; j < 2; ++j) {
#pragma unroll
            for (int r = 0; r < 8; ++r) {
                XW[(long)(mbase + r) * HDIM + n0 + j * 16 + ncol] =
                    f2bf(c[i][j][r]);
            }
        }
    }
}

// ---------------------------------------------------------------------------
// Phase 2: persistent recurrence kernel, ONE launch for all 512 steps.
//   h_{t+1} = tanh(XW_t + h_t @ W_hh)
// 256 blocks (one 16x16 output tile each), 8 waves = 8 K-chunks (split-K),
// LDS reduction, device-scope atomic grid barrier between steps.
// Each wave's W_hh tiles (4 x 1KB) are loaded ONCE and pinned in registers
// for all 512 steps: only the h-dependent A loads sit on the critical path.
// ---------------------------------------------------------------------------
__global__ __launch_bounds__(256)
void rnn_recurrence_kernel(const unsigned short* __restrict__ xw,   // [512*64,1024] bf16
                           const unsigned short* __restrict__ Whh,  // [1024,1024] bf16
                           unsigned short* __restrict__ h_a,        // [64,1024] bf16 (zeroed)
                           unsigned short* __restrict__ h_b,        // [64,1024] bf16
                           float* __restrict__ out_f32,             // [64,1024] f32
                           unsigned int* __restrict__ counter) {    // zeroed
    __shared__ float red[8 * 256];                                  // 8 KB

    const int tid   = threadIdx.x;
    const int lane  = tid & 31;
    const int wave  = tid >> 5;               // K-chunk id 0..7
    const int mtile = blockIdx.x & 3;
    const int n0    = (blockIdx.x >> 2) * 16;

    const int half  = lane >> 4;
    const int mrow  = lane & 15;
    const int kbase = wave * 128;

    const long a_off = (long)(mtile * 16 + mrow) * HDIM + kbase + half * 8;
    const unsigned short* brow = Whh + (long)(kbase + half * 8) * HDIM + n0 + mrow;

    // ---- loop-invariant: pin this wave's 4 W_hh tiles in registers ----
    ABPack b[4];
#pragma unroll
    for (int i = 0; i < 4; ++i) {
#pragma unroll
        for (int p = 0; p < 8; ++p)
            b[i].u32[p] = (unsigned)brow[(long)(i * 32 + 2 * p + (p >= 4 ? 8 : 0)) * HDIM] |
                          ((unsigned)brow[(long)(i * 32 + 2 * p + 1 + (p >= 4 ? 8 : 0)) * HDIM] << 16);
    }

    // reduction-side constants: thread t owns element (m = t/16, n = t%16)
    const int m    = tid >> 4;
    const int n    = tid & 15;
    const int ridx = (n + ((m >> 3) << 4)) * 8 + (m & 7);  // lane*8 + vgpr
    const long gidx = (long)(mtile * 16 + m) * HDIM + n0 + n;

    for (int t = 0; t < TSEQ; ++t) {
        const unsigned short* h_in  = (t & 1) ? h_b : h_a;
        unsigned short*       h_out = (t & 1) ? h_a : h_b;

        // A = previous hidden state (critical-path loads; one clause)
        const unsigned short* arow = h_in + a_off;
        ABPack a[4];
#pragma unroll
        for (int i = 0; i < 4; ++i) {
            a[i].q[0] = *(const uint4*)(arow + i * 32);
            a[i].q[1] = *(const uint4*)(arow + i * 32 + 16);
        }

        v8f c = {};
#pragma unroll
        for (int i = 0; i < 4; ++i) {
            c = __builtin_amdgcn_wmma_f32_16x16x32_bf16(
                false, a[i].v, false, b[i].v, (short)0, c, false, false);
        }

        // partial tile -> LDS (contiguous 32B per lane)
        {
            union { v8f v; uint4 q[2]; } cv;
            cv.v = c;
            float* slab = red + wave * 256 + lane * 8;
            *(uint4*)(slab)     = cv.q[0];
            *(uint4*)(slab + 4) = cv.q[1];
        }
        __syncthreads();

        // reduce 8 K-chunk partials, add XW_t, tanh, store next h
        float sum = 0.f;
#pragma unroll
        for (int w = 0; w < 8; ++w) sum += red[w * 256 + ridx];

        float pre = sum + bf2f(xw[(long)t * (BDIM * HDIM) + gidx]);
        float hv  = tanhf(pre);
        h_out[gidx] = f2bf(hv);
        if (t == TSEQ - 1) out_f32[gidx] = hv;

        // ---- grid-wide barrier: monotonic counter, release/acquire ----
        __threadfence();          // make h_out stores device-visible
        __syncthreads();          // also protects red[] reuse
        if (tid == 0) {
            atomicAdd(counter, 1u);
            const unsigned int target = (unsigned int)(t + 1) * NBLK;
            while (atomicAdd(counter, 0u) < target) {
                __builtin_amdgcn_s_sleep(2);
            }
        }
        __syncthreads();
        __threadfence();          // acquire: discard stale h lines
    }
}

// ---------------------------------------------------------------------------
// Launcher. Workspace layout (bytes):
//   [0,        2MB)   W_ih bf16
//   [2MB,      4MB)   W_hh bf16
//   [4MB,     68MB)   XW bf16  [32768 x 1024]
//   [68MB, +128KB)    h ping (bf16)
//   [.., +128KB)      h pong (bf16)
//   [.., +4B)         grid-barrier counter
// Total ~68.3 MB.
// ---------------------------------------------------------------------------
extern "C" void kernel_launch(void* const* d_in, const int* in_sizes, int n_in,
                              void* d_out, int out_size, void* d_ws, size_t ws_size,
                              hipStream_t stream) {
    const float* x    = (const float*)d_in[0];  // [512,64,1024]
    const float* w_ih = (const float*)d_in[1];  // [1024,1024]
    const float* w_hh = (const float*)d_in[2];  // [1024,1024]
    float* out = (float*)d_out;                 // [64,1024]

    char* ws = (char*)d_ws;
    unsigned short* wih_bf = (unsigned short*)(ws);
    unsigned short* whh_bf = (unsigned short*)(ws + (size_t)2 * 1024 * 1024);
    unsigned short* xw_bf  = (unsigned short*)(ws + (size_t)4 * 1024 * 1024);
    unsigned short* h_a    = (unsigned short*)(ws + (size_t)68 * 1024 * 1024);
    unsigned short* h_b    = h_a + BDIM * HDIM;
    unsigned int*   cnt    = (unsigned int*)(h_b + BDIM * HDIM);

    // weights -> bf16
    rnn_cvt_bf16_kernel<<<2048, 256, 0, stream>>>(w_ih, wih_bf, HDIM * HDIM);
    rnn_cvt_bf16_kernel<<<2048, 256, 0, stream>>>(w_hh, whh_bf, HDIM * HDIM);
    // h0 = 0, barrier counter = 0 (every call: deterministic, capture-safe)
    rnn_zero_kernel<<<128, 256, 0, stream>>>((unsigned int*)h_a, BDIM * HDIM / 2);
    rnn_zero_kernel<<<1, 32, 0, stream>>>(cnt, 1);

    // Phase 1: parallel input-projection GEMM for all timesteps
    rnn_xw_gemm_kernel<<<4096, 256, 0, stream>>>(x, wih_bf, xw_bf);

    // Phase 2: single persistent launch for the 512-step recurrence
    rnn_recurrence_kernel<<<NBLK, 256, 0, stream>>>(
        xw_bf, whh_bf, h_a, h_b, out, cnt);
}
